// CausalGraphGenerator_75179107550029
// MI455X (gfx1250) — matrix operand order repaired and checked by reading, not compiled, not run on hardware
//
#include <hip/hip_runtime.h>
#include <math.h>

typedef __attribute__((ext_vector_type(16))) _Float16 v16h;
typedef __attribute__((ext_vector_type(8)))  _Float16 v8h;
typedef __attribute__((ext_vector_type(8)))  float    v8f;

#define B_  4
#define N_  4096
#define H_  128
#define D_  64

// ---------------------------------------------------------------------------
// Pass 1: M1 = tanh(h @ W1 + b1), M2 = tanh(h @ W2 + b2), stored as f16.
// One 64-thread block per (b,n) row; h row staged in LDS; W accesses coalesced.
// ---------------------------------------------------------------------------
__global__ __launch_bounds__(64)
void proj_tanh_kernel(const float* __restrict__ h,
                      const float* __restrict__ W1, const float* __restrict__ b1,
                      const float* __restrict__ W2, const float* __restrict__ b2,
                      _Float16* __restrict__ M1, _Float16* __restrict__ M2) {
    __shared__ float sh[H_];
    const int row = blockIdx.x;            // b*N + n
    const int t = threadIdx.x;             // 0..63 (output feature d)
    const float* hrow = h + (size_t)row * H_;
    sh[t]      = hrow[t];
    sh[t + 64] = hrow[t + 64];
    __syncthreads();
    float a1 = b1[t], a2 = b2[t];
#pragma unroll 4
    for (int k = 0; k < H_; ++k) {
        float hv = sh[k];
        a1 += hv * W1[k * D_ + t];
        a2 += hv * W2[k * D_ + t];
    }
    M1[(size_t)row * D_ + t] = (_Float16)tanhf(a1);
    M2[(size_t)row * D_ + t] = (_Float16)tanhf(a2);
}

// ---------------------------------------------------------------------------
// WMMA fragment loaders (CDNA5 16x16x32 f16 layouts, wave32).
// A 16x32 (MxK): lanes 0-15 -> row M=lane, K {kc+0..7, kc+16..23}
//                lanes 16-31 -> row M=lane-16, K {kc+8..15, kc+24..31}
// B 32x16 (KxN): lane -> col N=lane&15, K {kc+(lane>>4)*8+0..7, kc+16+(lane>>4)*8+0..7}
// ---------------------------------------------------------------------------
__device__ inline v16h load_a_frag(const _Float16* __restrict__ base, int lane, int kc) {
    const int half = lane >> 4;
    const int r    = lane & 15;
    const _Float16* rowp = base + (size_t)r * D_;
    union { v16h v; v8h h[2]; } u;
    u.h[0] = *(const v8h*)(rowp + kc +      half * 8);
    u.h[1] = *(const v8h*)(rowp + kc + 16 + half * 8);
    return u.v;
}

__device__ inline v16h load_b_frag(const _Float16* __restrict__ base, int lane, int kc) {
    const int col = lane & 15;
    const int kh  = lane >> 4;
    const _Float16* rowp = base + (size_t)col * D_;
    union { v16h v; v8h h[2]; } u;
    u.h[0] = *(const v8h*)(rowp + kc +      kh * 8);
    u.h[1] = *(const v8h*)(rowp + kc + 16 + kh * 8);
    return u.v;
}

// Fast sigmoid: hardware rcp (v_rcp_f32) instead of IEEE-divide sequence.
__device__ inline float fast_sigmoid(float x) {
    return __builtin_amdgcn_rcpf(1.0f + __expf(-x));
}

// ---------------------------------------------------------------------------
// Pass 2: A[b,n,m] = sigmoid( (M1 M2^T)[n,m] - (M2 M1^T)[n,m] ), fused epilogue.
// 8 waves/block; each wave owns one 16-row strip and sweeps N/8 columns.
// ---------------------------------------------------------------------------
__global__ __launch_bounds__(256)
void score_kernel(const _Float16* __restrict__ M1, const _Float16* __restrict__ M2,
                  float* __restrict__ out) {
    const int lane = threadIdx.x & 31;
    const int wave = threadIdx.x >> 5;
    const int strip = blockIdx.x * 8 + wave;       // 0 .. B*(N/16)-1
    const int tilesPerRow = N_ / 16;               // 256
    const int b  = strip / tilesPerRow;
    const int it = strip % tilesPerRow;
    const int rowBase = it * 16;

    const size_t matOff = (size_t)b * N_ * D_;
    const _Float16* m1b = M1 + matOff;
    const _Float16* m2b = M2 + matOff;

    // A-fragments for this strip: held for the whole column sweep.
    const _Float16* aB1 = m1b + (size_t)rowBase * D_;
    const _Float16* aB2 = m2b + (size_t)rowBase * D_;
    const v16h a1k0 = load_a_frag(aB1, lane, 0);
    const v16h a1k1 = load_a_frag(aB1, lane, 32);
    const v16h a2k0 = load_a_frag(aB2, lane, 0);
    const v16h a2k1 = load_a_frag(aB2, lane, 32);

    float* outb = out + (size_t)b * N_ * N_;
    const int crow0 = rowBase + ((lane >> 4) << 3);
    const int ccol  = lane & 15;

    const int tilesPerChunk = tilesPerRow / 8;     // 32 column tiles per wave
    const int jt0 = blockIdx.y * tilesPerChunk;
    for (int jt = jt0; jt < jt0 + tilesPerChunk; ++jt) {
        const int colBase = jt * 16;
        const _Float16* bB1 = m2b + (size_t)colBase * D_;   // B for M1*M2^T
        const _Float16* bB2 = m1b + (size_t)colBase * D_;   // B for M2*M1^T
        const v16h b1k0 = load_b_frag(bB1, lane, 0);
        const v16h b1k1 = load_b_frag(bB1, lane, 32);
        const v16h b2k0 = load_b_frag(bB2, lane, 0);
        const v16h b2k1 = load_b_frag(bB2, lane, 32);

        v8f c1 = {}, c2 = {};
        c1 = __builtin_amdgcn_wmma_f32_16x16x32_f16(false, a1k0, false, b1k0, (short)0, c1, false, false);
        c1 = __builtin_amdgcn_wmma_f32_16x16x32_f16(false, a1k1, false, b1k1, (short)0, c1, false, false);
        c2 = __builtin_amdgcn_wmma_f32_16x16x32_f16(false, a2k0, false, b2k0, (short)0, c2, false, false);
        c2 = __builtin_amdgcn_wmma_f32_16x16x32_f16(false, a2k1, false, b2k1, (short)0, c2, false, false);

#pragma unroll
        for (int e = 0; e < 8; ++e) {
            const float s = fast_sigmoid(c1[e] - c2[e]);
            outb[(size_t)(crow0 + e) * N_ + colBase + ccol] = s;
        }
    }
}

// ---------------------------------------------------------------------------
// Pass 3: per-row exact top-k mask (lowest-index tie-break) + identity, in place.
// One 256-thread block per row; each thread owns 16 columns (coalesced).
// ---------------------------------------------------------------------------
__global__ __launch_bounds__(256)
void topk_kernel(float* __restrict__ out, const int* __restrict__ topk_ptr) {
    __shared__ float sval[256];
    __shared__ int   scol[256];
    const int row = blockIdx.x;             // b*N + n
    const int n   = row % N_;
    const int tid = threadIdx.x;
    float* rowp = out + (size_t)row * N_;

    float v[16], w[16];
#pragma unroll
    for (int s = 0; s < 16; ++s) {
        v[s] = rowp[s * 256 + tid];
        w[s] = v[s];
    }

    int k = *topk_ptr;
    if (k > N_) k = N_;
    unsigned keep = 0;

    for (int iter = 0; iter < k; ++iter) {
        // local argmax (value desc, index asc)
        float bv = -1.0f; int bc = 0x7fffffff;
#pragma unroll
        for (int s = 0; s < 16; ++s) {
            const int c = s * 256 + tid;
            if (w[s] > bv || (w[s] == bv && c < bc)) { bv = w[s]; bc = c; }
        }
        sval[tid] = bv; scol[tid] = bc;
        __syncthreads();
        for (int stride = 128; stride > 0; stride >>= 1) {
            if (tid < stride) {
                const float ov = sval[tid + stride]; const int oc = scol[tid + stride];
                if (ov > sval[tid] || (ov == sval[tid] && oc < scol[tid])) {
                    sval[tid] = ov; scol[tid] = oc;
                }
            }
            __syncthreads();
        }
        const int wc = scol[0];
        __syncthreads();            // protect scol[0] before next-iter overwrite
        if ((wc & 255) == tid) {    // owner removes winner, records keep bit
            const int s = wc >> 8;
            keep |= 1u << s;
            w[s] = -1.0f;
        }
    }

#pragma unroll
    for (int s = 0; s < 16; ++s) {
        const int c = s * 256 + tid;
        float val = ((keep >> s) & 1u) ? v[s] : 0.0f;
        if (c == n) val += 1.0f;    // + identity
        rowp[c] = val;
    }
}

// ---------------------------------------------------------------------------
extern "C" void kernel_launch(void* const* d_in, const int* in_sizes, int n_in,
                              void* d_out, int out_size, void* d_ws, size_t ws_size,
                              hipStream_t stream) {
    const float* h   = (const float*)d_in[0];
    const float* W1w = (const float*)d_in[1];
    const float* W1b = (const float*)d_in[2];
    const float* W2w = (const float*)d_in[3];
    const float* W2b = (const float*)d_in[4];
    const int*   tk  = (const int*)d_in[5];
    float* out = (float*)d_out;

    _Float16* M1 = (_Float16*)d_ws;                       // 2 MiB
    _Float16* M2 = M1 + (size_t)B_ * N_ * D_;             // 2 MiB

    proj_tanh_kernel<<<B_ * N_, 64, 0, stream>>>(h, W1w, W1b, W2w, W2b, M1, M2);

    dim3 g2(B_ * (N_ / 16) / 8, 8);                        // 128 x 8 blocks, 8192 waves
    score_kernel<<<g2, 256, 0, stream>>>(M1, M2, out);

    topk_kernel<<<B_ * N_, 256, 0, stream>>>(out, tk);
}
